// SurfaceConv2_15831249453211
// MI455X (gfx1250) — matrix operand order repaired and checked by reading, not compiled, not run on hardware
//
#include <hip/hip_runtime.h>
#include <hip/hip_bf16.h>

// ---------------------------------------------------------------------------
// CDNA5 (gfx1250) implementation of the SurfaceConv2 block.
// Core compute = bf16 WMMA GEMMs (v_wmma_f32_16x16x32_bf16) with fused
// per-channel BN-statistics epilogues (LDS atomics -> global atomics).
// Per-wave C tile = 32x64 (2x4 fragments) so each A fragment is reused by
// four WMMAs: 12 ds_load_b128 per K-step feed 8 v_wmma ops.
// ---------------------------------------------------------------------------

typedef __attribute__((ext_vector_type(16))) __bf16 v16bf;
typedef __attribute__((ext_vector_type(8)))  __bf16 v8bf;
typedef __attribute__((ext_vector_type(8)))  float  v8f;

#define BN_EPS 1e-5f
constexpr int Bsz    = 2;
constexpr int Npts   = 4096;
constexpr int NPOINT = 1024;
constexpr int Kn     = 16;
constexpr int C_PTS  = 61;
constexpr int IN_CH  = 64;   // 61 + 3
constexpr int PRE    = 32;
constexpr int OUT    = 64;
constexpr int Mrows  = Bsz * NPOINT * Kn;   // 32768
constexpr int Pcnt   = Bsz * NPOINT;        // 2048

// ---------------------------------------------------------------------------
__global__ void zero_f32_kernel(float* p, int n) {
  int i = blockIdx.x * blockDim.x + threadIdx.x;
  if (i < n) p[i] = 0.0f;
}

// new_xyz = gather(xyz, data_idx)  -> first 6144 floats of d_out
__global__ void newxyz_kernel(const float* __restrict__ xyz,
                              const int* __restrict__ didx,
                              float* __restrict__ out) {
  int i = blockIdx.x * blockDim.x + threadIdx.x;
  if (i >= Pcnt * 3) return;
  int c = i % 3;
  int pn = i / 3;                 // b*NPOINT + n
  int b = pn / NPOINT;
  out[i] = xyz[((size_t)b * Npts + didx[pn]) * 3 + c];
}

// A0[row][c] = concat(points, xyz) gathered by neighbor_lists; row = (b*NPOINT+n)*K+k
__global__ void gather_kernel(const float* __restrict__ xyz,
                              const float* __restrict__ points,
                              const int* __restrict__ nbr,
                              float* __restrict__ A0) {
  int i = blockIdx.x * blockDim.x + threadIdx.x;
  if (i >= Mrows * IN_CH) return;
  int c   = i & (IN_CH - 1);
  int row = i >> 6;
  int b   = row / (Kn * NPOINT);
  int idx = nbr[row];
  float v = (c < C_PTS) ? points[((size_t)b * Npts + idx) * C_PTS + c]
                        : xyz[((size_t)b * Npts + idx) * 3 + (c - C_PTS)];
  A0[i] = v;
}

// BN finalize: scale = g*rsqrt(var+eps), shift = b - mean*scale (training-mode stats)
__global__ void finalize_bn_kernel(const float* __restrict__ sum,
                                   const float* __restrict__ sqs,
                                   const float* __restrict__ gamma,
                                   const float* __restrict__ beta,
                                   float* __restrict__ scale,
                                   float* __restrict__ shift,
                                   float count, int nch) {
  int c = blockIdx.x * blockDim.x + threadIdx.x;
  if (c >= nch) return;
  float mean = sum[c] / count;
  float var  = sqs[c] / count - mean * mean;
  float sc   = gamma[c] * rsqrtf(var + BN_EPS);
  scale[c] = sc;
  shift[c] = beta[c] - mean * sc;
}

// ---------------------------------------------------------------------------
// Tiled bf16 WMMA GEMM:  C[M,N] = act(A[M,Kdim]) @ W[N,Kdim]^T + bias
//   act(x) = applyIn ? relu(x*inScale[k]+inShift[k]) : x   (folded BN+ReLU of
//   the previous layer, applied while staging A into LDS)
// Epilogue: writes C (f32 or bf16) and accumulates per-channel sum / sum^2
// (LDS reduction then global f32 atomics) for the next BN.
// Block = 256 threads = 8 waves (WAVES_M x 1); each wave computes an
// (FM*16) x (FN*16) C tile as FMxFN v_wmma_f32_16x16x32_bf16 fragments.
// FULLK=true strips the zero-pad scalar staging path (Kdim % 32 == 0).
// ---------------------------------------------------------------------------
template <int BM, int BN, int FM, int FN, bool FULLK, bool OUT_BF16>
__global__ __launch_bounds__(256) void gemm_wmma_kernel(
    const float* __restrict__ A, int Kdim,
    const float* __restrict__ inScale, const float* __restrict__ inShift, int applyIn,
    const float* __restrict__ W, const float* __restrict__ bias,
    float* __restrict__ Cf, __bf16* __restrict__ Cb,
    float* __restrict__ gSum, float* __restrict__ gSqs, int N) {
  constexpr int BK  = 32;
  constexpr int BKP = 40;  // padded stride (bf16 elems): 80B rows, 16B aligned
  static_assert(BM == 8 * FM * 16, "8 waves stacked along M");
  static_assert(BN == FN * 16, "one wave column");
  __shared__ __bf16 As[BM * BKP];
  __shared__ __bf16 Ws[BN * BKP];
  __shared__ float  sSum[BN];
  __shared__ float  sSqs[BN];

  const int tid  = threadIdx.x;
  const int n0   = blockIdx.x * BN;
  const int m0   = blockIdx.y * BM;
  const int wave = tid >> 5;
  const int lane = tid & 31;
  const int lm   = lane & 15;   // M (A) / N (B,C) index within 16
  const int hi   = lane >> 4;   // lane-half selects K/M groups per ISA layout
  const int wrow = wave * FM * 16;   // wave's row base within the block tile

  if (tid < BN) { sSum[tid] = 0.0f; sSqs[tid] = 0.0f; }

  v8f acc[FM][FN] = {};
  const int kSteps = (Kdim + BK - 1) / BK;

  for (int ks = 0; ks < kSteps; ++ks) {
    const int kk = ks * BK;

    // ---- stage A tile (with folded BN+ReLU of previous layer) ----
    for (int it = tid; it < BM * BK / 4; it += 256) {
      int r  = it / (BK / 4);
      int kq = (it % (BK / 4)) * 4;
      __bf16* dst = &As[r * BKP + kq];
      const float* src = A + (size_t)(m0 + r) * Kdim + kk + kq;
      if constexpr (FULLK) {
        const float4 f4 = *(const float4*)src;
        float vv[4] = {f4.x, f4.y, f4.z, f4.w};
#pragma unroll
        for (int u = 0; u < 4; ++u) {
          float v = vv[u];
          if (applyIn) {
            v = v * inScale[kk + kq + u] + inShift[kk + kq + u];
            v = v > 0.0f ? v : 0.0f;
          }
          dst[u] = (__bf16)v;
        }
      } else {
#pragma unroll
        for (int u = 0; u < 4; ++u) {
          int k = kk + kq + u;
          float v = 0.0f;
          if (k < Kdim) {
            v = src[u];
            if (applyIn) { v = v * inScale[k] + inShift[k]; v = v > 0.0f ? v : 0.0f; }
          }
          dst[u] = (__bf16)v;
        }
      }
    }
    // ---- stage W tile ----
    for (int it = tid; it < BN * BK / 4; it += 256) {
      int r  = it / (BK / 4);
      int kq = (it % (BK / 4)) * 4;
      __bf16* dst = &Ws[r * BKP + kq];
      const float* src = W + (size_t)(n0 + r) * Kdim + kk + kq;
      if constexpr (FULLK) {
        const float4 f4 = *(const float4*)src;
        dst[0] = (__bf16)f4.x; dst[1] = (__bf16)f4.y;
        dst[2] = (__bf16)f4.z; dst[3] = (__bf16)f4.w;
      } else {
#pragma unroll
        for (int u = 0; u < 4; ++u) {
          int k = kk + kq + u;
          dst[u] = (__bf16)((k < Kdim) ? src[u] : 0.0f);
        }
      }
    }
    // prefetch next K slab of A while we compute on this one
    if (ks + 1 < kSteps) {
      const float* nxt = A + (size_t)(m0 + (tid % BM)) * Kdim + kk + BK;
      __builtin_prefetch(nxt, 0, 1);
    }
    __syncthreads();

    // ---- build fragments per ISA 16-bit A/B VGPR layouts, two b128 LDS
    //      reads per fragment ----
    union FragU { v16bf v; v8bf h[2]; };
    FragU fa[FM], fb[FN];
#pragma unroll
    for (int i = 0; i < FM; ++i) {
      const __bf16* base = &As[(wrow + i * 16 + lm) * BKP];
      fa[i].h[0] = *(const v8bf*)(base + 8 * hi);        // K = 8*hi + 0..7
      fa[i].h[1] = *(const v8bf*)(base + 16 + 8 * hi);   // K = 16 + 8*hi + 0..7
    }
#pragma unroll
    for (int j = 0; j < FN; ++j) {
      const __bf16* base = &Ws[(j * 16 + lm) * BKP];
      fb[j].h[0] = *(const v8bf*)(base + 8 * hi);        // K = 8*hi + 0..7
      fb[j].h[1] = *(const v8bf*)(base + 16 + 8 * hi);   // K = 16 + 8*hi + 0..7
    }
#pragma unroll
    for (int i = 0; i < FM; ++i)
#pragma unroll
      for (int j = 0; j < FN; ++j)
        acc[i][j] = __builtin_amdgcn_wmma_f32_16x16x32_bf16(
            /*neg_a=*/false, fa[i].v, /*neg_b=*/false, fb[j].v,
            /*c_mod=*/(short)0, acc[i][j], /*reuse_a=*/false, /*reuse_b=*/false);
    __syncthreads();
  }

  // ---- epilogue: bias, store, per-channel sum / sum^2 ----
#pragma unroll
  for (int i = 0; i < FM; ++i) {
#pragma unroll
    for (int j = 0; j < FN; ++j) {
      const int colL = j * 16 + lm;   // column within block tile
      const int col  = n0 + colL;
      const float bv = bias[col];
      float ls = 0.0f, lq = 0.0f;
#pragma unroll
      for (int r = 0; r < 8; ++r) {
        const int m = m0 + wrow + i * 16 + 8 * hi + r;  // C layout: M = r + 8*hi
        float v = acc[i][j][r] + bv;
        ls += v; lq += v * v;
        const size_t idx = (size_t)m * N + col;
        if constexpr (OUT_BF16) Cb[idx] = (__bf16)v; else Cf[idx] = v;
      }
      atomicAdd(&sSum[colL], ls);
      atomicAdd(&sSqs[colL], lq);
    }
  }
  __syncthreads();
  if (tid < BN) {
    atomicAdd(&gSum[n0 + tid], sSum[tid]);
    atomicAdd(&gSqs[n0 + tid], sSqs[tid]);
  }
}

// ---------------------------------------------------------------------------
// new_points_pre[pt,o] = sum_{k,p} relu(aff3(H3[pt*K+k, p*64+o])) *
//                                  relu(aff2(C2[pt*K+k, p]))
// One 64-thread block per point; g staged in LDS; H3 reads are 128B coalesced.
// ---------------------------------------------------------------------------
__global__ __launch_bounds__(64) void aggregate_kernel(
    const __bf16* __restrict__ H3, const float* __restrict__ s3, const float* __restrict__ t3,
    const float* __restrict__ C2, const float* __restrict__ s2, const float* __restrict__ t2,
    float* __restrict__ NP) {
  __shared__ float gl[Kn * PRE];  // 512 floats
  const int pt = blockIdx.x;      // 0..2047
  const int o  = threadIdx.x;     // 0..63
  for (int idx = o; idx < Kn * PRE; idx += 64) {
    int k = idx / PRE, p = idx % PRE;
    float v = C2[(size_t)(pt * Kn + k) * PRE + p];
    v = v * s2[p] + t2[p];
    gl[idx] = v > 0.0f ? v : 0.0f;
  }
  __syncthreads();
  float acc = 0.0f;
  for (int k = 0; k < Kn; ++k) {
    const __bf16* hrow = H3 + (size_t)(pt * Kn + k) * (PRE * OUT);
#pragma unroll
    for (int p = 0; p < PRE; ++p) {
      const int c = p * OUT + o;
      float h = (float)hrow[c];
      h = h * s3[c] + t3[c];
      h = h > 0.0f ? h : 0.0f;
      acc += h * gl[k * PRE + p];
    }
  }
  NP[(size_t)pt * OUT + o] = acc;
}

__global__ void out_affine_relu_kernel(const float* __restrict__ Lpre,
                                       const float* __restrict__ sc,
                                       const float* __restrict__ sh,
                                       float* __restrict__ out) {
  int i = blockIdx.x * blockDim.x + threadIdx.x;
  if (i >= Pcnt * OUT) return;
  int c = i % OUT;
  float v = Lpre[i] * sc[c] + sh[c];
  out[i] = v > 0.0f ? v : 0.0f;
}

// ---------------------------------------------------------------------------
extern "C" void kernel_launch(void* const* d_in, const int* in_sizes, int n_in,
                              void* d_out, int out_size, void* d_ws, size_t ws_size,
                              hipStream_t stream) {
  (void)in_sizes; (void)n_in; (void)out_size; (void)ws_size;
  // ---- inputs (setup_inputs dict order; params flattened list-by-list) ----
  const float* xyz    = (const float*)d_in[0];
  const float* points = (const float*)d_in[1];
  const float* lcoord = (const float*)d_in[2];
  const int*   nbr    = (const int*)d_in[3];
  const int*   didx   = (const int*)d_in[4];
  const float* conv_w[2]  = {(const float*)d_in[5],  (const float*)d_in[6]};
  const float* conv_b[2]  = {(const float*)d_in[7],  (const float*)d_in[8]};
  const float* bn_g[2]    = {(const float*)d_in[9],  (const float*)d_in[10]};
  const float* bn_b[2]    = {(const float*)d_in[11], (const float*)d_in[12]};
  const float* wn_w[4]    = {(const float*)d_in[13], (const float*)d_in[14],
                             (const float*)d_in[15], (const float*)d_in[16]};
  const float* wn_b[4]    = {(const float*)d_in[17], (const float*)d_in[18],
                             (const float*)d_in[19], (const float*)d_in[20]};
  const float* wn_bn_g[4] = {(const float*)d_in[21], (const float*)d_in[22],
                             (const float*)d_in[23], (const float*)d_in[24]};
  const float* wn_bn_b[4] = {(const float*)d_in[25], (const float*)d_in[26],
                             (const float*)d_in[27], (const float*)d_in[28]};
  const float* lin_w = (const float*)d_in[29];
  const float* lin_b = (const float*)d_in[30];
  const float* bnl_g = (const float*)d_in[31];
  const float* bnl_b = (const float*)d_in[32];

  float* out_xyz = (float*)d_out;                 // 2*1024*3 = 6144
  float* out_np  = (float*)d_out + Pcnt * 3;      // 2*1024*64 = 131072

  // ---- workspace layout (float offsets) ----
  float* ws = (float*)d_ws;
  // stats blocks: [sum | sqs | scale | shift] per BN layer
  const int nch[7] = {32, 32, 256, 256, 512, 2048, 64};  // c0,c1,w0,w1,w2,w3,lin
  float* stat[7];
  size_t off = 0;
  for (int l = 0; l < 7; ++l) { stat[l] = ws + off; off += (size_t)4 * nch[l]; }
  // -> off == 12800 (multiple of 16 floats)
  auto sSum = [&](int l) { return stat[l]; };
  auto sSqs = [&](int l) { return stat[l] + nch[l]; };
  auto sScl = [&](int l) { return stat[l] + 2 * nch[l]; };
  auto sSft = [&](int l) { return stat[l] + 3 * nch[l]; };
  float* A0 = ws + off;  off += (size_t)Mrows * IN_CH;   // 32768x64
  float* C1 = ws + off;  off += (size_t)Mrows * PRE;     // 32768x32
  float* C2 = ws + off;  off += (size_t)Mrows * PRE;
  float* H0 = ws + off;  off += (size_t)Mrows * 256;
  float* H1 = ws + off;  off += (size_t)Mrows * 256;
  float* H2 = ws + off;  off += (size_t)Mrows * 512;
  __bf16* H3 = (__bf16*)(ws + off); off += ((size_t)Mrows * 2048) / 2;  // bf16
  float* NPagg = ws + off; off += (size_t)Pcnt * OUT;
  float* Lpre  = ws + off; off += (size_t)Pcnt * OUT;

  const int statsTotal = 12800;
  zero_f32_kernel<<<(statsTotal + 255) / 256, 256, 0, stream>>>(ws, statsTotal);
  newxyz_kernel<<<(Pcnt * 3 + 255) / 256, 256, 0, stream>>>(xyz, didx, out_xyz);
  gather_kernel<<<(Mrows * IN_CH + 255) / 256, 256, 0, stream>>>(xyz, points, nbr, A0);

  // ---- conv stack (N=32 -> 256x32 block, 2x2 frags/wave) ----
  gemm_wmma_kernel<256, 32, 2, 2, true, false><<<dim3(1, Mrows / 256), 256, 0, stream>>>(
      A0, IN_CH, nullptr, nullptr, 0, conv_w[0], conv_b[0],
      C1, nullptr, sSum(0), sSqs(0), 32);
  finalize_bn_kernel<<<1, 32, 0, stream>>>(sSum(0), sSqs(0), bn_g[0], bn_b[0],
                                           sScl(0), sSft(0), (float)Mrows, 32);
  gemm_wmma_kernel<256, 32, 2, 2, true, false><<<dim3(1, Mrows / 256), 256, 0, stream>>>(
      C1, PRE, sScl(0), sSft(0), 1, conv_w[1], conv_b[1],
      C2, nullptr, sSum(1), sSqs(1), 32);
  finalize_bn_kernel<<<1, 32, 0, stream>>>(sSum(1), sSqs(1), bn_g[1], bn_b[1],
                                           sScl(1), sSft(1), (float)Mrows, 32);

  // ---- WeightNet: 3->256->256->512->2048 (256x64 block, 2x4 frags/wave) ----
  gemm_wmma_kernel<256, 64, 2, 4, false, false><<<dim3(256 / 64, Mrows / 256), 256, 0, stream>>>(
      lcoord, 3, nullptr, nullptr, 0, wn_w[0], wn_b[0],
      H0, nullptr, sSum(2), sSqs(2), 256);
  finalize_bn_kernel<<<1, 256, 0, stream>>>(sSum(2), sSqs(2), wn_bn_g[0], wn_bn_b[0],
                                            sScl(2), sSft(2), (float)Mrows, 256);
  gemm_wmma_kernel<256, 64, 2, 4, true, false><<<dim3(256 / 64, Mrows / 256), 256, 0, stream>>>(
      H0, 256, sScl(2), sSft(2), 1, wn_w[1], wn_b[1],
      H1, nullptr, sSum(3), sSqs(3), 256);
  finalize_bn_kernel<<<1, 256, 0, stream>>>(sSum(3), sSqs(3), wn_bn_g[1], wn_bn_b[1],
                                            sScl(3), sSft(3), (float)Mrows, 256);
  gemm_wmma_kernel<256, 64, 2, 4, true, false><<<dim3(512 / 64, Mrows / 256), 256, 0, stream>>>(
      H1, 256, sScl(3), sSft(3), 1, wn_w[2], wn_b[2],
      H2, nullptr, sSum(4), sSqs(4), 512);
  finalize_bn_kernel<<<2, 256, 0, stream>>>(sSum(4), sSqs(4), wn_bn_g[2], wn_bn_b[2],
                                            sScl(4), sSft(4), (float)Mrows, 512);
  gemm_wmma_kernel<256, 64, 2, 4, true, true><<<dim3(2048 / 64, Mrows / 256), 256, 0, stream>>>(
      H2, 512, sScl(4), sSft(4), 1, wn_w[3], wn_b[3],
      nullptr, H3, sSum(5), sSqs(5), 2048);
  finalize_bn_kernel<<<8, 256, 0, stream>>>(sSum(5), sSqs(5), wn_bn_g[3], wn_bn_b[3],
                                            sScl(5), sSft(5), (float)Mrows, 2048);

  // ---- weighted aggregation over (k, p) ----
  aggregate_kernel<<<Pcnt, 64, 0, stream>>>(H3, sScl(5), sSft(5),
                                            C2, sScl(1), sSft(1), NPagg);

  // ---- final linear (2048x64 @ 64x64) + BN stats ----
  gemm_wmma_kernel<256, 64, 2, 4, true, false><<<dim3(1, Pcnt / 256), 256, 0, stream>>>(
      NPagg, OUT, nullptr, nullptr, 0, lin_w, lin_b,
      Lpre, nullptr, sSum(6), sSqs(6), OUT);
  finalize_bn_kernel<<<1, 64, 0, stream>>>(sSum(6), sSqs(6), bnl_g, bnl_b,
                                           sScl(6), sSft(6), (float)Pcnt, OUT);
  out_affine_relu_kernel<<<(Pcnt * OUT + 255) / 256, 256, 0, stream>>>(
      Lpre, sScl(6), sSft(6), out_np);
}
